// DeformableAttention_50190987821709
// MI455X (gfx1250) — matrix operand order repaired and checked by reading, not compiled, not run on hardware
//
#include <hip/hip_runtime.h>
#include <math.h>

typedef __attribute__((ext_vector_type(16))) __bf16 v16bf;
typedef __attribute__((ext_vector_type(8)))  float  v8f;

union Frag16 { v16bf v; uint4 u[2]; };

__device__ __forceinline__ unsigned short f2bf(float f) {
  unsigned int u = __float_as_uint(f);
  u += 0x7FFFu + ((u >> 16) & 1u);   // round-to-nearest-even
  return (unsigned short)(u >> 16);
}

#define WMMA_BF16(A, B, C) \
  __builtin_amdgcn_wmma_f32_16x16x32_bf16(false, (A), false, (B), (short)0, (C), false, false)

// ---------------------------------------------------------------------------
// Generic per-batch GEMM: C[z][M,N] = W[M,K] * X[z][K,N] (+ bias[M])
// fp32 in/out, bf16 WMMA inner. M,N multiples of 64; K multiple of 32.
// Block = 128 threads (4 waves), 64x64 output tile, wave owns a 16x64 strip.
// ---------------------------------------------------------------------------
__global__ __launch_bounds__(128)
void gemm_wx_bf16(const float* __restrict__ W, const float* __restrict__ X,
                  float* __restrict__ C, const float* __restrict__ bias,
                  int M, int N, int K)
{
  __shared__ __align__(16) unsigned short As[64][32];  // [m][k]
  __shared__ __align__(16) unsigned short Bs[64][32];  // [n][k]  (X transposed)

  const int tid  = threadIdx.x;
  const int lane = tid & 31;
  const int wave = tid >> 5;
  const int n0   = blockIdx.x * 64;
  const int m0   = blockIdx.y * 64;
  X += (size_t)blockIdx.z * K * N;
  C += (size_t)blockIdx.z * M * N;

  v8f acc0 = {}, acc1 = {}, acc2 = {}, acc3 = {};

  const int am = tid >> 1;            // 0..63  (A row)
  const int ak = (tid & 1) * 16;      // 0,16   (A k half)
  const int bk = tid >> 2;            // 0..31  (B k)
  const int bn = (tid & 3) * 16;      // 0..48  (B n quarter)

  const int mrow = wave * 16 + (lane & 15);
  const int kbA  = (lane >> 4) * 16;  // A frag contiguous base
  const int kbB  = (lane >> 4) * 16;  // B frag contiguous base
  const int nc   = lane & 15;

  for (int kk = 0; kk < K; kk += 32) {
    // stage A tile 64x32 (k contiguous per thread)
    {
      const float4* wp = (const float4*)(W + (size_t)(m0 + am) * K + kk + ak);
      unsigned short* arow = &As[am][ak];
#pragma unroll
      for (int i = 0; i < 4; ++i) {
        float4 t = wp[i];
        arow[4*i+0] = f2bf(t.x); arow[4*i+1] = f2bf(t.y);
        arow[4*i+2] = f2bf(t.z); arow[4*i+3] = f2bf(t.w);
      }
    }
    // stage B tile 32x64 transposed -> Bs[n][k]
    {
      const float4* xp = (const float4*)(X + (size_t)(kk + bk) * N + n0 + bn);
#pragma unroll
      for (int i = 0; i < 4; ++i) {
        float4 t = xp[i];
        Bs[bn+4*i+0][bk] = f2bf(t.x); Bs[bn+4*i+1][bk] = f2bf(t.y);
        Bs[bn+4*i+2][bk] = f2bf(t.z); Bs[bn+4*i+3][bk] = f2bf(t.w);
      }
    }
    __syncthreads();

    Frag16 a;
    a.u[0] = *(const uint4*)&As[mrow][kbA];
    a.u[1] = *(const uint4*)&As[mrow][kbA + 8];

    Frag16 bf;
    bf.u[0] = *(const uint4*)&Bs[nc][kbB];      bf.u[1] = *(const uint4*)&Bs[nc][kbB + 8];
    acc0 = WMMA_BF16(a.v, bf.v, acc0);
    bf.u[0] = *(const uint4*)&Bs[16+nc][kbB];   bf.u[1] = *(const uint4*)&Bs[16+nc][kbB + 8];
    acc1 = WMMA_BF16(a.v, bf.v, acc1);
    bf.u[0] = *(const uint4*)&Bs[32+nc][kbB];   bf.u[1] = *(const uint4*)&Bs[32+nc][kbB + 8];
    acc2 = WMMA_BF16(a.v, bf.v, acc2);
    bf.u[0] = *(const uint4*)&Bs[48+nc][kbB];   bf.u[1] = *(const uint4*)&Bs[48+nc][kbB + 8];
    acc3 = WMMA_BF16(a.v, bf.v, acc3);
    __syncthreads();
  }

  const int hi8 = (lane >> 4) * 8;   // C/D row half per ISA layout
#pragma unroll
  for (int r = 0; r < 8; ++r) {
    const int mr = m0 + wave * 16 + hi8 + r;
    const float bv = bias ? bias[mr] : 0.0f;
    float* crow = C + (size_t)mr * N + n0 + nc;
    crow[ 0] = acc0[r] + bv;
    crow[16] = acc1[r] + bv;
    crow[32] = acc2[r] + bv;
    crow[48] = acc3[r] + bv;
  }
}

// ---------------------------------------------------------------------------
// Offsets: depthwise 1x1 (stride-4 downsample) -> exact GELU -> 1x1 -> tanh*4
// off layout: [bg=64][2][256]
// ---------------------------------------------------------------------------
__global__ __launch_bounds__(256)
void offsets_kernel(const float* __restrict__ q,
                    const float* __restrict__ w1, const float* __restrict__ b1,
                    const float* __restrict__ w2, float* __restrict__ off)
{
  const int idx = blockIdx.x * 256 + threadIdx.x;  // 64 bg * 256 pos
  const int bg = idx >> 8, pos = idx & 255;
  const int gy = pos >> 4, gx = pos & 15;
  const int b = bg >> 3, g = bg & 7;
  float a0 = 0.0f, a1 = 0.0f;
  for (int d = 0; d < 64; ++d) {
    const float v = q[((size_t)b * 512 + g * 64 + d) * 4096 + (gy * 4) * 64 + gx * 4];
    const float t = v * w1[d] + b1[d];
    const float ge = 0.5f * t * (1.0f + erff(t * 0.70710678118654752f));
    a0 += w2[d] * ge;
    a1 += w2[64 + d] * ge;
  }
  off[(size_t)bg * 512 + pos]       = tanhf(a0) * 4.0f;
  off[(size_t)bg * 512 + 256 + pos] = tanhf(a1) * 4.0f;
}

// ---------------------------------------------------------------------------
// Bilinear grid-sample (zeros padding, align_corners=False) with the
// reference's swapped (h,w)->(x,y) coordinate convention.
// kvf layout: [b][512][256]
// ---------------------------------------------------------------------------
__global__ __launch_bounds__(256)
void gridsample_kernel(const float* __restrict__ q, const float* __restrict__ off,
                       float* __restrict__ kvf)
{
  const int idx = blockIdx.x * 256 + threadIdx.x;   // 64 bg * 64 d * 256 pos
  const int pos = idx & 255;
  const int d   = (idx >> 8) & 63;
  const int bg  = idx >> 14;
  const int gy = pos >> 4, gxc = pos & 15;
  const int b = bg >> 3, g = bg & 7;

  const float o0 = off[(size_t)bg * 512 + pos];
  const float o1 = off[(size_t)bg * 512 + 256 + pos];
  const float nh = 2.0f * ((float)gy  + o0) * (1.0f / 15.0f) - 1.0f;  // -> x coord
  const float nw = 2.0f * ((float)gxc + o1) * (1.0f / 15.0f) - 1.0f;  // -> y coord
  const float gxi = ((nh + 1.0f) * 64.0f - 1.0f) * 0.5f;
  const float gyi = ((nw + 1.0f) * 64.0f - 1.0f) * 0.5f;
  const float x0f = floorf(gxi), y0f = floorf(gyi);
  const int x0 = (int)x0f, y0 = (int)y0f;
  const float wx1 = gxi - x0f, wx0 = 1.0f - wx1;
  const float wy1 = gyi - y0f, wy0 = 1.0f - wy1;

  const float* img = q + ((size_t)b * 512 + g * 64 + d) * 4096;
  const bool xv0 = (x0 >= 0) & (x0 < 64), xv1 = (x0 + 1 >= 0) & (x0 + 1 < 64);
  const bool yv0 = (y0 >= 0) & (y0 < 64), yv1 = (y0 + 1 >= 0) & (y0 + 1 < 64);
  const float v00 = (yv0 & xv0) ? img[y0 * 64 + x0]           : 0.0f;
  const float v01 = (yv0 & xv1) ? img[y0 * 64 + x0 + 1]       : 0.0f;
  const float v10 = (yv1 & xv0) ? img[(y0 + 1) * 64 + x0]     : 0.0f;
  const float v11 = (yv1 & xv1) ? img[(y0 + 1) * 64 + x0 + 1] : 0.0f;

  kvf[((size_t)b * 512 + g * 64 + d) * 256 + pos] =
      v00 * wy0 * wx0 + v01 * wy0 * wx1 + v10 * wy1 * wx0 + v11 * wy1 * wx1;
}

// ---------------------------------------------------------------------------
// Flash-style attention per (64-query strip, head, batch).
// q: [b][512][4096] (unscaled), kv: [b][1024][256] (k = ch 0..511, v = 512..1023)
// out: [b][512][4096]
// ---------------------------------------------------------------------------
__global__ __launch_bounds__(128)
void deform_attn(const float* __restrict__ q, const float* __restrict__ kv,
                 float* __restrict__ out)
{
  __shared__ __align__(16) unsigned short Qs[64][64];      // [query][dim], scaled bf16
  __shared__ __align__(16) unsigned short Ks[32][64];      // [key][dim]
  __shared__ __align__(16) unsigned short Vs[64][32];      // [dim][key]
  __shared__ __align__(16) unsigned short Pls[4][16][32];  // per-wave P bounce

  const int tid = threadIdx.x, lane = tid & 31, wave = tid >> 5;
  const int pix0 = blockIdx.x * 64;
  const int h = blockIdx.y;
  const int b = blockIdx.z;
  const float scale = 0.125f;   // 64^-0.5

  // stage scaled Q strip
  {
    const int p = tid & 63;
    for (int d = tid >> 6; d < 64; d += 2)
      Qs[p][d] = f2bf(q[((size_t)b * 512 + h * 64 + d) * 4096 + pix0 + p] * scale);
  }
  __syncthreads();

  const int mrow = wave * 16 + (lane & 15);
  const int kbA  = (lane >> 4) * 16;
  const int kbB  = (lane >> 4) * 16;
  const int nc   = lane & 15;
  const int hi8  = (lane >> 4) * 8;

  Frag16 q0, q1;   // Q fragments for dims 0..31 and 32..63, kept in registers
  q0.u[0] = *(const uint4*)&Qs[mrow][kbA];       q0.u[1] = *(const uint4*)&Qs[mrow][kbA + 8];
  q1.u[0] = *(const uint4*)&Qs[mrow][32 + kbA];  q1.u[1] = *(const uint4*)&Qs[mrow][32 + kbA + 8];

  v8f O0 = {}, O1 = {}, O2 = {}, O3 = {};
  float rmax[8], ssum[8];
#pragma unroll
  for (int r = 0; r < 8; ++r) { rmax[r] = -1e30f; ssum[r] = 0.0f; }

  const int ld_d = tid >> 1;
  const int ld_k = (tid & 1) * 16;
  const float* kbase = kv + ((size_t)b * 1024 + h * 64 + ld_d) * 256;
  const float* vbase = kv + ((size_t)b * 1024 + 512 + h * 64 + ld_d) * 256;

  for (int kc = 0; kc < 256; kc += 32) {
    __syncthreads();
    {
      const float4* kp = (const float4*)(kbase + kc + ld_k);
      const float4* vp = (const float4*)(vbase + kc + ld_k);
      unsigned short* vrow = &Vs[ld_d][ld_k];
#pragma unroll
      for (int i = 0; i < 4; ++i) {
        float4 t = kp[i];
        Ks[ld_k + 4*i + 0][ld_d] = f2bf(t.x); Ks[ld_k + 4*i + 1][ld_d] = f2bf(t.y);
        Ks[ld_k + 4*i + 2][ld_d] = f2bf(t.z); Ks[ld_k + 4*i + 3][ld_d] = f2bf(t.w);
        float4 u = vp[i];
        vrow[4*i+0] = f2bf(u.x); vrow[4*i+1] = f2bf(u.y);
        vrow[4*i+2] = f2bf(u.z); vrow[4*i+3] = f2bf(u.w);
      }
    }
    __syncthreads();

    // sim tiles S0 (keys kc..kc+15), S1 (keys kc+16..kc+31); K dim = 64 = 2 WMMA steps
    v8f S0 = {}, S1 = {};
    {
      Frag16 kf;
      kf.u[0] = *(const uint4*)&Ks[nc][kbB];           kf.u[1] = *(const uint4*)&Ks[nc][kbB + 8];
      S0 = WMMA_BF16(q0.v, kf.v, S0);
      kf.u[0] = *(const uint4*)&Ks[nc][32 + kbB];      kf.u[1] = *(const uint4*)&Ks[nc][32 + kbB + 8];
      S0 = WMMA_BF16(q1.v, kf.v, S0);
      kf.u[0] = *(const uint4*)&Ks[16 + nc][kbB];      kf.u[1] = *(const uint4*)&Ks[16 + nc][kbB + 8];
      S1 = WMMA_BF16(q0.v, kf.v, S1);
      kf.u[0] = *(const uint4*)&Ks[16 + nc][32 + kbB]; kf.u[1] = *(const uint4*)&Ks[16 + nc][32 + kbB + 8];
      S1 = WMMA_BF16(q1.v, kf.v, S1);
    }

    // online softmax update (row stats live replicated across each 16-lane half)
#pragma unroll
    for (int r = 0; r < 8; ++r) {
      float tm = fmaxf(S0[r], S1[r]);
      tm = fmaxf(tm, __shfl_xor(tm, 1, 16));
      tm = fmaxf(tm, __shfl_xor(tm, 2, 16));
      tm = fmaxf(tm, __shfl_xor(tm, 4, 16));
      tm = fmaxf(tm, __shfl_xor(tm, 8, 16));
      const float nm   = fmaxf(rmax[r], tm);
      const float resc = __expf(rmax[r] - nm);
      const float p0 = __expf(S0[r] - nm);
      const float p1 = __expf(S1[r] - nm);
      float ps = p0 + p1;
      ps += __shfl_xor(ps, 1, 16);
      ps += __shfl_xor(ps, 2, 16);
      ps += __shfl_xor(ps, 4, 16);
      ps += __shfl_xor(ps, 8, 16);
      ssum[r] = ssum[r] * resc + ps;
      rmax[r] = nm;
      O0[r] *= resc; O1[r] *= resc; O2[r] *= resc; O3[r] *= resc;
      Pls[wave][hi8 + r][nc]      = f2bf(p0);
      Pls[wave][hi8 + r][16 + nc] = f2bf(p1);
    }
    // wave-private C-layout -> A-layout bounce through LDS
    asm volatile("s_wait_dscnt 0" ::: "memory");

    Frag16 pf;
    pf.u[0] = *(const uint4*)&Pls[wave][lane & 15][kbA];
    pf.u[1] = *(const uint4*)&Pls[wave][lane & 15][kbA + 8];

    Frag16 vf;
    vf.u[0] = *(const uint4*)&Vs[nc][kbB];      vf.u[1] = *(const uint4*)&Vs[nc][kbB + 8];
    O0 = WMMA_BF16(pf.v, vf.v, O0);
    vf.u[0] = *(const uint4*)&Vs[16 + nc][kbB]; vf.u[1] = *(const uint4*)&Vs[16 + nc][kbB + 8];
    O1 = WMMA_BF16(pf.v, vf.v, O1);
    vf.u[0] = *(const uint4*)&Vs[32 + nc][kbB]; vf.u[1] = *(const uint4*)&Vs[32 + nc][kbB + 8];
    O2 = WMMA_BF16(pf.v, vf.v, O2);
    vf.u[0] = *(const uint4*)&Vs[48 + nc][kbB]; vf.u[1] = *(const uint4*)&Vs[48 + nc][kbB + 8];
    O3 = WMMA_BF16(pf.v, vf.v, O3);
  }

  // normalize and store: out[b][h*64 + dim][pixel]
#pragma unroll
  for (int r = 0; r < 8; ++r) {
    const float inv = 1.0f / ssum[r];
    const int prow = pix0 + wave * 16 + hi8 + r;
    out[((size_t)b * 512 + h * 64 +  0 + nc) * 4096 + prow] = O0[r] * inv;
    out[((size_t)b * 512 + h * 64 + 16 + nc) * 4096 + prow] = O1[r] * inv;
    out[((size_t)b * 512 + h * 64 + 32 + nc) * 4096 + prow] = O2[r] * inv;
    out[((size_t)b * 512 + h * 64 + 48 + nc) * 4096 + prow] = O3[r] * inv;
  }
}

// ---------------------------------------------------------------------------
extern "C" void kernel_launch(void* const* d_in, const int* in_sizes, int n_in,
                              void* d_out, int out_size, void* d_ws, size_t ws_size,
                              hipStream_t stream)
{
  const float* x      = (const float*)d_in[0];
  const float* w_q    = (const float*)d_in[1];
  const float* w_off1 = (const float*)d_in[2];
  const float* b_off1 = (const float*)d_in[3];
  const float* w_off2 = (const float*)d_in[4];
  const float* w_kv   = (const float*)d_in[5];
  const float* w_out  = (const float*)d_in[6];
  const float* b_out  = (const float*)d_in[7];
  float* out = (float*)d_out;
  float* ws  = (float*)d_ws;

  // workspace layout (floats): total ~140 MB
  float* q_ws    = ws;                     // 8*512*4096   = 16,777,216
  float* off_ws  = q_ws   + 16777216;      // 64*2*256     =     32,768
  float* kvf_ws  = off_ws + 32768;         // 8*512*256    =  1,048,576
  float* kv_ws   = kvf_ws + 1048576;       // 8*1024*256   =  2,097,152
  float* attn_ws = kv_ws  + 2097152;       // 8*512*4096   = 16,777,216

  // q = w_q @ x                                  [8][512,4096]
  gemm_wx_bf16<<<dim3(64, 8, 8), 128, 0, stream>>>(w_q, x, q_ws, nullptr, 512, 4096, 512);
  // offsets from downsampled q                   [64][2][256]
  offsets_kernel<<<dim3(64), 256, 0, stream>>>(q_ws, w_off1, b_off1, w_off2, off_ws);
  // deformable sampling                          [8][512][256]
  gridsample_kernel<<<dim3(4096), 256, 0, stream>>>(q_ws, off_ws, kvf_ws);
  // kv = w_kv @ kv_feats                         [8][1024,256]
  gemm_wx_bf16<<<dim3(4, 16, 8), 128, 0, stream>>>(w_kv, kvf_ws, kv_ws, nullptr, 1024, 256, 512);
  // attention (flash-style, WMMA)                [8][512][4096]
  deform_attn<<<dim3(64, 8, 8), 128, 0, stream>>>(q_ws, kv_ws, attn_ws);
  // out = w_out @ attn + b_out                   [8][512,4096]
  gemm_wx_bf16<<<dim3(64, 8, 8), 128, 0, stream>>>(w_out, attn_ws, out, b_out, 512, 4096, 512);
}
